// LazyAttention_7404523618935
// MI455X (gfx1250) — matrix operand order repaired and checked by reading, not compiled, not run on hardware
//
#include <hip/hip_runtime.h>

#define B_   32
#define LQ_  512
#define LK_  512
#define D_   1024

typedef __attribute__((ext_vector_type(16))) __bf16 v16bf;
typedef __attribute__((ext_vector_type(8)))  __bf16 v8bf;
typedef __attribute__((ext_vector_type(8)))  float  v8f;

__device__ __forceinline__ v16bf make_frag(const __bf16* lo_p, const __bf16* hi_p) {
  v8bf lo = *(const v8bf*)lo_p;
  v8bf hi = *(const v8bf*)hi_p;
  return __builtin_shufflevector(lo, hi, 0,1,2,3,4,5,6,7,8,9,10,11,12,13,14,15);
}

// Branch-free tanh: 1 - 2*rcp(exp(2x)+1).  exp->inf => 1, exp->0 => -1.
// v_exp_f32 + v_rcp_f32 only; result feeds f32->bf16-grade output, 1ulp rcp ok.
__device__ __forceinline__ float fast_tanh(float x) {
  float t = __expf(2.0f * x);
  return 1.0f - 2.0f * __builtin_amdgcn_rcpf(t + 1.0f);
}

// ---------------------------------------------------------------------------
// Prep: f32 -> bf16 grid-stride convert
// ---------------------------------------------------------------------------
__global__ void cvt_f32_bf16(const float* __restrict__ s, __bf16* __restrict__ d, int n) {
  int i = blockIdx.x * blockDim.x + threadIdx.x;
  int stride = gridDim.x * blockDim.x;
  for (; i < n; i += stride) d[i] = (__bf16)s[i];
}

// Prep: ctxT[b][d][lk] = context[b][lk][d], bf16 out, LDS-tiled transpose
__global__ void transpose_ctx_bf16(const float* __restrict__ src, __bf16* __restrict__ dst) {
  __shared__ float tile[32][33];
  const int b  = blockIdx.z;
  const int d0 = blockIdx.x * 32;   // column in src (D)
  const int l0 = blockIdx.y * 32;   // row in src (LK)
  const float* s = src + (long long)b * LK_ * D_;
  __bf16* dd = dst + (long long)b * D_ * LK_;
  for (int i = threadIdx.y; i < 32; i += 8)
    tile[i][threadIdx.x] = s[(long long)(l0 + i) * D_ + d0 + threadIdx.x];
  __syncthreads();
  for (int i = threadIdx.y; i < 32; i += 8)
    dd[(long long)(d0 + i) * LK_ + l0 + threadIdx.x] = (__bf16)tile[threadIdx.x][i];
}

// ---------------------------------------------------------------------------
// Generic bf16 WMMA GEMM:  C[m][n] = sum_k A[m][k] * Bt[n][k]
// Block tile 128x128 (256 threads = 8 waves), wave tile 32x64 (2x4 WMMA tiles)
// K-step 64.  Staging: 8 distinct short-lived uint4 temps, loads issued above
// the barrier (8-deep MLP, no cross-compute liveness -> no scratch spills).
// Latency hiding across blocks: ~110 VGPRs + 36KB LDS => multiple blocks/WGP.
// LDS row stride 72 bf16 (144B = 36 banks) -> conflict-free ds_load_b128.
// ---------------------------------------------------------------------------
#define LSTR 72

__global__ __launch_bounds__(256) void gemm_bf16_wmma(
    const __bf16* __restrict__ A, int lda, long long sA,
    const __bf16* __restrict__ Bt, int ldb, long long sB,
    float* __restrict__ Cf, __bf16* __restrict__ Cb, int ldc, long long sC,
    int K, int do_tanh)
{
  __shared__ __attribute__((aligned(16))) __bf16 As[128 * LSTR];
  __shared__ __attribute__((aligned(16))) __bf16 Bs[128 * LSTR];
  const int tid  = threadIdx.x;
  const int wid  = tid >> 5, lane = tid & 31;
  const int lm   = lane & 15, kg = lane >> 4;
  const long long bm = (long long)blockIdx.x * 128;
  const long long bn = (long long)blockIdx.y * 128;
  A  += (long long)blockIdx.z * sA;
  Bt += (long long)blockIdx.z * sB;
  const long long cbase = (long long)blockIdx.z * sC;

  const int wm = (wid >> 1) * 32;   // wave row offset: 0/32/64/96
  const int wn = (wid & 1) * 64;    // wave col offset: 0/64

  const int ldrow = tid >> 1;          // 0..127
  const int ldseg = (tid & 1) * 32;    // 0 or 32 (bf16 elems)

  const uint4* ga = (const uint4*)(A  + (bm + ldrow) * (long long)lda + ldseg);
  const uint4* gb = (const uint4*)(Bt + (bn + ldrow) * (long long)ldb + ldseg);
  const int gstep = (64 * 2) / 16;     // 64 bf16 = 8 uint4 per K-step
  uint4* sa = (uint4*)&As[ldrow * LSTR + ldseg];
  uint4* sb = (uint4*)&Bs[ldrow * LSTR + ldseg];

  v8f acc[2][4];
  #pragma unroll
  for (int mt = 0; mt < 2; ++mt)
    #pragma unroll
    for (int nt = 0; nt < 4; ++nt) acc[mt][nt] = 0.0f;

  for (int k0 = 0; k0 < K; k0 += 64) {
    // 8 independent global loads in flight before the barrier; temps die at
    // the LDS stores right after it (short live ranges -> stay in VGPRs).
    uint4 a0 = ga[0], a1 = ga[1], a2 = ga[2], a3 = ga[3];
    uint4 b0 = gb[0], b1 = gb[1], b2 = gb[2], b3 = gb[3];
    ga += gstep; gb += gstep;
    __syncthreads();               // previous tile's fragment reads done
    sa[0] = a0; sa[1] = a1; sa[2] = a2; sa[3] = a3;
    sb[0] = b0; sb[1] = b1; sb[2] = b2; sb[3] = b3;
    __syncthreads();               // tile visible to all waves

    #pragma unroll
    for (int kk = 0; kk < 64; kk += 32) {
      v16bf af[2], bf[4];
      #pragma unroll
      for (int mt = 0; mt < 2; ++mt) {
        // A 16x32 frag: elems 0..7 = K kg*8.. ; elems 8..15 = K 16+kg*8..
        const __bf16* pa = &As[(wm + mt * 16 + lm) * LSTR + kk + kg * 8];
        af[mt] = make_frag(pa, pa + 16);
      }
      #pragma unroll
      for (int nt = 0; nt < 4; ++nt) {
        // B 32x16 frag: elems 0..7 = K kg*8.. ; elems 8..15 = K 16+kg*8..
        const __bf16* pb = &Bs[(wn + nt * 16 + lm) * LSTR + kk + kg * 8];
        bf[nt] = make_frag(pb, pb + 16);
      }
      #pragma unroll
      for (int nt = 0; nt < 4; ++nt)
        #pragma unroll
        for (int mt = 0; mt < 2; ++mt)
          acc[mt][nt] = __builtin_amdgcn_wmma_f32_16x16x32_bf16(
              false, af[mt], false, bf[nt], (short)0, acc[mt][nt], false, false);
    }
  }

  #pragma unroll
  for (int mt = 0; mt < 2; ++mt)
    #pragma unroll
    for (int nt = 0; nt < 4; ++nt)
      #pragma unroll
      for (int j = 0; j < 8; ++j) {
        const long long row = bm + wm + mt * 16 + kg * 8 + j;
        const long long col = bn + wn + nt * 16 + lm;
        float v = acc[mt][nt][j];
        if (do_tanh) v = fast_tanh(v);
        const long long idx = cbase + row * (long long)ldc + col;
        if (Cf) Cf[idx] = v;
        if (Cb) Cb[idx] = (__bf16)v;
      }
}

// ---------------------------------------------------------------------------
// Fused scores + mask + softmax + mask:
//   one block = (batch b, 16 query rows), full LK=512 width (8 waves x 64 cols)
//   scores[m][n] = q[m][:] . ctx[n][:]  via WMMA, A/B frags straight from global
// ---------------------------------------------------------------------------
__global__ __launch_bounds__(256) void attn_scores_softmax(
    const __bf16* __restrict__ combined,   // q bf16 at column offset 1024, ld 2048
    const __bf16* __restrict__ ctx,        // [B, LK, D] bf16
    const int* __restrict__ q_lens, const int* __restrict__ c_lens,
    float* __restrict__ w_f32,             // [B, LQ, LK] (d_out region)
    __bf16* __restrict__ w_bf)             // [B, LQ, LK] (workspace)
{
  __shared__ float red[8][16];
  __shared__ float rowstat[16];
  const int blk = blockIdx.x;
  const int b   = blk >> 5;          // 32 row-blocks per batch
  const int m0  = (blk & 31) * 16;
  const int tid = threadIdx.x;
  const int wid = tid >> 5, lane = tid & 31;
  const int lm  = lane & 15, kg = lane >> 4;

  const __bf16* qb = combined + ((long long)(b * LQ_ + m0)) * 2048 + 1024;
  const __bf16* cb = ctx + (long long)b * LK_ * D_;

  v8f acc[4];
  #pragma unroll
  for (int nt = 0; nt < 4; ++nt) acc[nt] = 0.0f;

  for (int k0 = 0; k0 < D_; k0 += 32) {
    const __bf16* pa = qb + (long long)lm * 2048 + k0 + kg * 8;
    v16bf a = make_frag(pa, pa + 16);
    v16bf bf[4];
    #pragma unroll
    for (int nt = 0; nt < 4; ++nt) {
      const int n = wid * 64 + nt * 16 + lm;
      const __bf16* pb = cb + (long long)n * D_ + k0 + kg * 8;
      bf[nt] = make_frag(pb, pb + 16);
    }
    #pragma unroll
    for (int nt = 0; nt < 4; ++nt)
      acc[nt] = __builtin_amdgcn_wmma_f32_16x16x32_bf16(
          false, a, false, bf[nt], (short)0, acc[nt], false, false);
  }

  const int qlen = q_lens[b], clen = c_lens[b];

  // mask (-1e9 like reference NEG) + per-row max
  float pval[8];
  #pragma unroll
  for (int j = 0; j < 8; ++j) {
    const int row = m0 + kg * 8 + j;
    const bool qpad = row >= qlen;
    float mj = -3.0e38f;
    #pragma unroll
    for (int nt = 0; nt < 4; ++nt) {
      const int col = wid * 64 + nt * 16 + lm;
      float v = (qpad || col >= clen) ? -1.0e9f : acc[nt][j];
      acc[nt][j] = v;
      mj = fmaxf(mj, v);
    }
    pval[j] = mj;
  }
  #pragma unroll
  for (int off = 1; off < 16; off <<= 1)
    #pragma unroll
    for (int j = 0; j < 8; ++j)
      pval[j] = fmaxf(pval[j], __shfl_xor(pval[j], off, 32));

  if (lane == 0 || lane == 16) {
    #pragma unroll
    for (int j = 0; j < 8; ++j) red[wid][kg * 8 + j] = pval[j];
  }
  __syncthreads();
  if (tid < 16) {
    float m = red[0][tid];
    #pragma unroll
    for (int w = 1; w < 8; ++w) m = fmaxf(m, red[w][tid]);
    rowstat[tid] = m;
  }
  __syncthreads();
  float rowmax[8];
  #pragma unroll
  for (int j = 0; j < 8; ++j) rowmax[j] = rowstat[kg * 8 + j];
  __syncthreads();

  // exp + per-row sum
  #pragma unroll
  for (int j = 0; j < 8; ++j) {
    float s = 0.0f;
    #pragma unroll
    for (int nt = 0; nt < 4; ++nt) {
      float e = __expf(acc[nt][j] - rowmax[j]);
      acc[nt][j] = e;
      s += e;
    }
    pval[j] = s;
  }
  #pragma unroll
  for (int off = 1; off < 16; off <<= 1)
    #pragma unroll
    for (int j = 0; j < 8; ++j)
      pval[j] += __shfl_xor(pval[j], off, 32);

  if (lane == 0 || lane == 16) {
    #pragma unroll
    for (int j = 0; j < 8; ++j) red[wid][kg * 8 + j] = pval[j];
  }
  __syncthreads();
  if (tid < 16) {
    float s = 0.0f;
    #pragma unroll
    for (int w = 0; w < 8; ++w) s += red[w][tid];
    rowstat[tid] = s;
  }
  __syncthreads();

  // normalize, post-softmax zeroing, emit f32 (output) + bf16 (for mix GEMM)
  #pragma unroll
  for (int j = 0; j < 8; ++j) {
    const int row = m0 + kg * 8 + j;
    const bool qpad = row >= qlen;
    const float inv = __builtin_amdgcn_rcpf(rowstat[kg * 8 + j]);
    #pragma unroll
    for (int nt = 0; nt < 4; ++nt) {
      const int col = wid * 64 + nt * 16 + lm;
      const float wv = (qpad || col >= clen) ? 0.0f : acc[nt][j] * inv;
      const long long idx = ((long long)(b * LQ_ + row)) * LK_ + col;
      w_f32[idx] = wv;
      w_bf[idx]  = (__bf16)wv;
    }
  }
}

// ---------------------------------------------------------------------------
// Launch
// ---------------------------------------------------------------------------
extern "C" void kernel_launch(void* const* d_in, const int* in_sizes, int n_in,
                              void* d_out, int out_size, void* d_ws, size_t ws_size,
                              hipStream_t stream) {
  const float* query   = (const float*)d_in[0];
  const float* context = (const float*)d_in[1];
  const float* W_in    = (const float*)d_in[2];
  const float* W_out   = (const float*)d_in[3];
  const int*   q_lens  = (const int*)d_in[4];
  const int*   c_lens  = (const int*)d_in[5];

  float* out   = (float*)d_out;
  float* w_out = out + (long long)B_ * LQ_ * D_;   // second output tensor

  char* ws = (char*)d_ws;
  __bf16* combined = (__bf16*)(ws);                         // [16384, 2048]  64 MB (mix | q)
  __bf16* ctx_bf   = (__bf16*)(ws + 67108864LL);            // [B, LK, D]     32 MB
  __bf16* ctxT_bf  = (__bf16*)(ws + 100663296LL);           // [B, D, LK]     32 MB
  __bf16* w_bf     = (__bf16*)(ws + 134217728LL);           // [B, LQ, LK]    16 MB
  __bf16* q_bf     = (__bf16*)(ws + 150994944LL);           // [16384, 1024]  32 MB
  __bf16* win_bf   = (__bf16*)(ws + 184549376LL);           // [1024, 1024]    2 MB
  __bf16* wout_bf  = (__bf16*)(ws + 186646528LL);           // [1024, 2048]    4 MB

  // prep: bf16 conversions + context transpose
  cvt_f32_bf16<<<2048, 256, 0, stream>>>(query,   q_bf,    B_ * LQ_ * D_);
  cvt_f32_bf16<<<2048, 256, 0, stream>>>(context, ctx_bf,  B_ * LK_ * D_);
  cvt_f32_bf16<<<1024, 256, 0, stream>>>(W_in,    win_bf,  D_ * D_);
  cvt_f32_bf16<<<1024, 256, 0, stream>>>(W_out,   wout_bf, D_ * 2 * D_);
  transpose_ctx_bf16<<<dim3(D_ / 32, LK_ / 32, B_), dim3(32, 8), 0, stream>>>(context, ctxT_bf);

  // GEMM1: q = query @ W_in^T  -> combined[:, 1024:2048]
  gemm_bf16_wmma<<<dim3(16384 / 128, 1024 / 128, 1), 256, 0, stream>>>(
      q_bf, D_, 0LL, win_bf, D_, 0LL,
      nullptr, combined + 1024, 2048, 0LL, D_, 0);

  // scores + mask + softmax + mask -> w (f32 to d_out, bf16 to ws)
  attn_scores_softmax<<<dim3(B_ * LQ_ / 16), 256, 0, stream>>>(
      combined, ctx_bf, q_lens, c_lens, w_out, w_bf);

  // GEMM3 (batched): mix = w @ ctx  -> combined[:, 0:1024]
  gemm_bf16_wmma<<<dim3(LQ_ / 128, 1024 / 128, B_), 256, 0, stream>>>(
      w_bf, LK_, (long long)LQ_ * LK_,
      ctxT_bf, LK_, (long long)D_ * LK_,
      nullptr, combined, 2048, (long long)LQ_ * 2048, LK_, 0);

  // GEMM4: out = tanh(combined @ W_out^T)
  gemm_bf16_wmma<<<dim3(16384 / 128, 1024 / 128, 1), 256, 0, stream>>>(
      combined, 2 * D_, 0LL, wout_bf, 2 * D_, 0LL,
      out, nullptr, D_, 0LL, 2 * D_, 1);
}
